// Int8Linear_39848706573719
// MI455X (gfx1250) — matrix operand order repaired and checked by reading, not compiled, not run on hardware
//
#include <hip/hip_runtime.h>
#include <hip/hip_bf16.h>

typedef _Float16 v16h __attribute__((ext_vector_type(16)));
typedef _Float16 v8h  __attribute__((ext_vector_type(8)));
typedef _Float16 v4h  __attribute__((ext_vector_type(4)));
typedef float    v8f  __attribute__((ext_vector_type(8)));

#define LS 40  // LDS row stride in halves (32 + 8 pad) -> 80B rows, 16B-aligned chunks

// ---------------- kernel 1: zero the absmax cell ----------------
__global__ void k_init(unsigned* absmax_bits) {
    if (threadIdx.x == 0) *absmax_bits = 0u;
}

// ---------------- kernel 2: abs-max over W (bit-pattern atomicMax) ----------------
__global__ void k_absmax(const float* __restrict__ W, size_t n, unsigned* absmax_bits) {
    __shared__ unsigned red[256];
    unsigned m = 0u;
    size_t stride = (size_t)gridDim.x * blockDim.x;
    for (size_t i = (size_t)blockIdx.x * blockDim.x + threadIdx.x; i < n; i += stride) {
        unsigned b = __float_as_uint(fabsf(W[i]));
        m = (b > m) ? b : m;
    }
    red[threadIdx.x] = m;
    __syncthreads();
    for (int s = 128; s > 0; s >>= 1) {
        if ((int)threadIdx.x < s) {
            unsigned o = red[threadIdx.x + s];
            if (o > red[threadIdx.x]) red[threadIdx.x] = o;
        }
        __syncthreads();
    }
    if (threadIdx.x == 0) atomicMax(absmax_bits, red[0]);
}

// ---------------- kernel 3a: quantize W -> f16 integer mantissas (exact in f16) ----------------
__global__ void k_quant_w(const float* __restrict__ W, _Float16* __restrict__ wh,
                          size_t n4, const unsigned* __restrict__ absmax_bits) {
    float amax = __uint_as_float(*absmax_bits);
    float r = (amax > 0.0f) ? (127.0f / amax) : 0.0f;  // 1/scale
    size_t i4 = (size_t)blockIdx.x * blockDim.x + threadIdx.x;
    if (i4 >= n4) return;
    const float4 v = ((const float4*)W)[i4];
    v4h o;
    float q;
    q = rintf(v.x * r); q = fminf(127.0f, fmaxf(-127.0f, q)); o.x = (_Float16)q;
    q = rintf(v.y * r); q = fminf(127.0f, fmaxf(-127.0f, q)); o.y = (_Float16)q;
    q = rintf(v.z * r); q = fminf(127.0f, fmaxf(-127.0f, q)); o.z = (_Float16)q;
    q = rintf(v.w * r); q = fminf(127.0f, fmaxf(-127.0f, q)); o.w = (_Float16)q;
    ((v4h*)wh)[i4] = o;
}

// ---------------- kernel 3b: convert x -> f16 ----------------
__global__ void k_cvt_x(const float* __restrict__ x, _Float16* __restrict__ xh, size_t n4) {
    size_t i4 = (size_t)blockIdx.x * blockDim.x + threadIdx.x;
    if (i4 >= n4) return;
    const float4 v = ((const float4*)x)[i4];
    v4h o;
    o.x = (_Float16)v.x; o.y = (_Float16)v.y; o.z = (_Float16)v.z; o.w = (_Float16)v.w;
    ((v4h*)xh)[i4] = o;
}

// ---------------- kernel 4: WMMA GEMM  y[t,o] = scale * sum_k xh[t,k]*wh[o,k] + b[o] ----------------
// grid: (N/128, T/128), block: 256 (8 waves). Each wave: 16 rows x 128 cols.
// Double-buffered LDS fed by global_load_async_to_lds_b128 (ASYNCcnt). All 8 B fragments
// are pinned live simultaneously so the 8 WMMAs issue back-to-back behind one ds-wait.
__global__ __launch_bounds__(256)
void k_gemm(const _Float16* __restrict__ xh, const _Float16* __restrict__ wh,
            const float* __restrict__ bias, float* __restrict__ out,
            const unsigned* __restrict__ absmax_bits, int K, int N) {
    __shared__ _Float16 sX[2][128 * LS];
    __shared__ _Float16 sW[2][128 * LS];

    const int tid    = threadIdx.x;
    const int waveId = tid >> 5;
    const int lane   = tid & 31;
    const int m      = lane & 15;          // row/col within fragment
    const bool hi    = lane >= 16;

    const int rowBase = blockIdx.y * 128;  // rows of x / out
    const int colBase = blockIdx.x * 128;  // out features

    // this thread's two staging chunks (each chunk = 8 halves = 16B)
    const int c0row  = tid >> 2;           // chunks 0..255: rows 0..63
    const int c0col  = (tid & 3) * 8;
    const int c1row  = c0row + 64;         // chunks 256..511: rows 64..127

    // wave-relative LDS byte offsets of this thread's destinations
    const unsigned ldsX0 = (unsigned)(uintptr_t)&sX[0][c0row * LS + c0col];
    const unsigned ldsX1 = (unsigned)(uintptr_t)&sX[0][c1row * LS + c0col];
    const unsigned ldsW0 = (unsigned)(uintptr_t)&sW[0][c0row * LS + c0col];
    const unsigned ldsW1 = (unsigned)(uintptr_t)&sW[0][c1row * LS + c0col];
    const unsigned bufStride = (unsigned)(128 * LS * sizeof(_Float16));

    // A-fragment K offsets (halves): lanes0-15 K in {0..15}; lanes16-31 {16..31}
    const int aK0 = hi ? 16 : 0;
    // B-fragment: lane holds one output column, 16 contiguous K halves
    const int bK0 = hi ? 16 : 0;

    v8f acc[8] = {};

    const int nIter = K >> 5;  // K/32

    // async issue of one K-tile pair into buffer `buf` (4 instructions -> ASYNCcnt +4)
    #define ISSUE_TILE(buf, k0)                                                          \
        do {                                                                             \
            unsigned bo = (unsigned)(buf) * bufStride;                                   \
            const _Float16* gx0 = xh + (size_t)(rowBase + c0row) * K + (k0) + c0col;     \
            const _Float16* gx1 = xh + (size_t)(rowBase + c1row) * K + (k0) + c0col;     \
            const _Float16* gw0 = wh + (size_t)(colBase + c0row) * K + (k0) + c0col;     \
            const _Float16* gw1 = wh + (size_t)(colBase + c1row) * K + (k0) + c0col;     \
            asm volatile("global_load_async_to_lds_b128 %0, %1, off"                     \
                         :: "v"(ldsX0 + bo), "v"((unsigned long long)gx0) : "memory");   \
            asm volatile("global_load_async_to_lds_b128 %0, %1, off"                     \
                         :: "v"(ldsX1 + bo), "v"((unsigned long long)gx1) : "memory");   \
            asm volatile("global_load_async_to_lds_b128 %0, %1, off"                     \
                         :: "v"(ldsW0 + bo), "v"((unsigned long long)gw0) : "memory");   \
            asm volatile("global_load_async_to_lds_b128 %0, %1, off"                     \
                         :: "v"(ldsW1 + bo), "v"((unsigned long long)gw1) : "memory");   \
        } while (0)

    ISSUE_TILE(0, 0);  // prologue

    for (int it = 0; it < nIter; ++it) {
        const int buf = it & 1;

        if (it + 1 < nIter) {
            ISSUE_TILE(buf ^ 1, (it + 1) << 5);               // overlap next tile's DMA
            asm volatile("s_wait_asynccnt 0x4" ::: "memory"); // current tile's 4 loads done
        } else {
            asm volatile("s_wait_asynccnt 0x0" ::: "memory");
        }
        __syncthreads();  // all waves' async data for `buf` visible

        // ---- A fragment for this wave's 16 rows ----
        const _Float16* aRow = &sX[buf][(waveId * 16 + m) * LS];
        v8h alo = *(const v8h*)(aRow + aK0);
        v8h ahi = *(const v8h*)(aRow + aK0 + 8);
        v16h a = __builtin_shufflevector(alo, ahi, 0,1,2,3,4,5,6,7,8,9,10,11,12,13,14,15);

        // ---- load all 8 B fragments, pin them live, then 8 back-to-back WMMAs ----
        const _Float16* bBase = &sW[buf][m * LS + bK0];
        v16h bf0, bf1, bf2, bf3, bf4, bf5, bf6, bf7;
        #define LOADB(i, dst)                                                             \
            do {                                                                          \
                v8h lo = *(const v8h*)(bBase + (i) * 16 * LS);                            \
                v8h h2 = *(const v8h*)(bBase + (i) * 16 * LS + 8);                        \
                dst = __builtin_shufflevector(lo, h2, 0,1,2,3,4,5,6,7,8,9,10,11,12,13,14,15); \
            } while (0)
        LOADB(0, bf0); LOADB(1, bf1); LOADB(2, bf2); LOADB(3, bf3);
        LOADB(4, bf4); LOADB(5, bf5); LOADB(6, bf6); LOADB(7, bf7);
        #undef LOADB
        // empty asm pins all fragments (and A) live simultaneously:
        // one ds-wait here, then the WMMA burst issues with no LDS stalls between ops
        asm volatile("" : "+v"(a), "+v"(bf0), "+v"(bf1), "+v"(bf2), "+v"(bf3),
                          "+v"(bf4), "+v"(bf5), "+v"(bf6), "+v"(bf7));

        acc[0] = __builtin_amdgcn_wmma_f32_16x16x32_f16(false, a, false, bf0, (short)0, acc[0], false, false);
        acc[1] = __builtin_amdgcn_wmma_f32_16x16x32_f16(false, a, false, bf1, (short)0, acc[1], false, false);
        acc[2] = __builtin_amdgcn_wmma_f32_16x16x32_f16(false, a, false, bf2, (short)0, acc[2], false, false);
        acc[3] = __builtin_amdgcn_wmma_f32_16x16x32_f16(false, a, false, bf3, (short)0, acc[3], false, false);
        acc[4] = __builtin_amdgcn_wmma_f32_16x16x32_f16(false, a, false, bf4, (short)0, acc[4], false, false);
        acc[5] = __builtin_amdgcn_wmma_f32_16x16x32_f16(false, a, false, bf5, (short)0, acc[5], false, false);
        acc[6] = __builtin_amdgcn_wmma_f32_16x16x32_f16(false, a, false, bf6, (short)0, acc[6], false, false);
        acc[7] = __builtin_amdgcn_wmma_f32_16x16x32_f16(false, a, false, bf7, (short)0, acc[7], false, false);

        __syncthreads();  // safe to overwrite `buf` next iteration
    }
    #undef ISSUE_TILE

    // ---- epilogue: y = acc*scale + bias ----
    const float scale = __uint_as_float(*absmax_bits) * (1.0f / 127.0f);
    const int rAdd = hi ? 8 : 0;
    #pragma unroll
    for (int nt = 0; nt < 8; ++nt) {
        int col = colBase + nt * 16 + m;
        float bv = bias[col];
        #pragma unroll
        for (int r = 0; r < 8; ++r) {
            int row = rowBase + waveId * 16 + r + rAdd;
            out[(size_t)row * N + col] = acc[nt][r] * scale + bv;
        }
    }
}

extern "C" void kernel_launch(void* const* d_in, const int* in_sizes, int n_in,
                              void* d_out, int out_size, void* d_ws, size_t ws_size,
                              hipStream_t stream) {
    const float* x = (const float*)d_in[0];
    const float* W = (const float*)d_in[1];
    const float* b = (const float*)d_in[2];
    float* out = (float*)d_out;

    const size_t nW = (size_t)in_sizes[1];                      // out*in
    const int    N  = in_sizes[2];                              // out features
    const int    K  = (int)(nW / (size_t)N);                    // in features
    const int    T  = (int)((size_t)in_sizes[0] / (size_t)K);   // batch rows
    const size_t nX = (size_t)in_sizes[0];

    // workspace layout: [absmax u32 @0] [xh : T*K halves @256] [wh : N*K halves]
    unsigned*  absmax = (unsigned*)d_ws;
    _Float16*  xh     = (_Float16*)((char*)d_ws + 256);
    _Float16*  wh     = xh + nX;

    k_init<<<1, 1, 0, stream>>>(absmax);

    k_absmax<<<1024, 256, 0, stream>>>(W, nW, absmax);

    {
        size_t n4 = nW / 4;
        k_quant_w<<<(unsigned)((n4 + 255) / 256), 256, 0, stream>>>(W, wh, n4, absmax);
    }
    {
        size_t n4 = nX / 4;
        k_cvt_x<<<(unsigned)((n4 + 255) / 256), 256, 0, stream>>>(x, xh, n4);
    }

    dim3 grid(N / 128, T / 128);
    k_gemm<<<grid, 256, 0, stream>>>(xh, wh, b, out, absmax, K, N);
}
